// MultiHeadSelfAttention_4389456576642
// MI455X (gfx1250) — matrix operand order repaired and checked by reading, not compiled, not run on hardware
//
#include <hip/hip_runtime.h>
#include <hip/hip_bf16.h>

// ---------------------------------------------------------------------------
// MHSA on gfx1250 via v_wmma_f32_16x16x32_f16 (f16 in, f32 accumulate).
// Pipeline: [cvt/transposes] -> [QKV gemm, async-LDS double-buffered]
//           -> [flash attention, async-LDS k/v tiles] -> [out proj]
// Fragment loads are batched ahead of WMMA bursts with sched_barrier(0) so the
// waitcnt pass emits staggered partial waits instead of full serialization.
// ---------------------------------------------------------------------------

#define HEADS   16
#define ADIM    64
#define DM      1024
#define BATCH_  8
#define SEQ_    1024
#define ROWS    (BATCH_ * SEQ_)   // 8192

typedef _Float16 half8   __attribute__((ext_vector_type(8)));
typedef _Float16 half16  __attribute__((ext_vector_type(16)));
typedef float    floatx8 __attribute__((ext_vector_type(8)));

#define DEV static __device__ __forceinline__

DEV half16 cat8(half8 lo, half8 hi) {
    half16 r;
#pragma unroll
    for (int i = 0; i < 8; ++i) { r[i] = lo[i]; r[i + 8] = hi[i]; }
    return r;
}

// A fragment (16x32, f16) from row-major [rows x ld(halves)] at (row0, k0).
DEV half16 load_a(const _Float16* p, int ld, int row0, int k0, int m, int hw) {
    const _Float16* base = p + (size_t)(row0 + m) * ld + k0 + hw * 8;
    half8 lo = *(const half8*)(base);
    half8 hi = *(const half8*)(base + 16);
    return cat8(lo, hi);
}

// B fragment (32x16 K-by-N, f16), B stored TRANSPOSED as [N][K] row-major.
DEV half16 load_b(const _Float16* p, int ld, int n0, int k0, int m, int hw) {
    const _Float16* base = p + (size_t)(n0 + m) * ld + k0 + hw * 8;
    half8 lo = *(const half8*)(base);
    half8 hi = *(const half8*)(base + 16);
    return cat8(lo, hi);
}

DEV floatx8 wmma_f16(half16 a, half16 b, floatx8 c) {
    return __builtin_amdgcn_wmma_f32_16x16x32_f16(
        false, a, false, b, (short)0, c, false, false);
}

// Keep batched fragment loads from being re-sunk next to their consumers.
DEV void sched_fence() { __builtin_amdgcn_sched_barrier(0); }

// CDNA5 async copy: global -> LDS, 16 bytes per lane, tracked by ASYNCcnt.
DEV void async_copy16(unsigned lds_addr, const _Float16* g) {
    asm volatile("global_load_async_to_lds_b128 %0, %1, off"
                 :: "v"(lds_addr), "v"((unsigned long long)(uintptr_t)g)
                 : "memory");
}
DEV void wait_async0() {
    asm volatile("s_wait_asynccnt 0x0" ::: "memory");
}

// ---------------------------------------------------------------------------
// Prep kernels
// ---------------------------------------------------------------------------
__global__ void cvt_x_kernel(const float* __restrict__ x, _Float16* __restrict__ x16, int n) {
    int i = blockIdx.x * blockDim.x + threadIdx.x;
    if (i < n) x16[i] = (_Float16)x[i];
}

__global__ void transpose_w_kernel(const float* __restrict__ W, _Float16* __restrict__ Wt) {
    int i = blockIdx.x * blockDim.x + threadIdx.x;
    if (i >= HEADS * 3 * ADIM * DM) return;
    int d  = i & (DM - 1);
    int a  = (i >> 10) & (ADIM - 1);
    int hp = i >> 16;                       // h*3+p
    Wt[i] = (_Float16)W[(size_t)hp * DM * ADIM + (size_t)d * ADIM + a];
}

__global__ void transpose_wz_kernel(const float* __restrict__ Wz, _Float16* __restrict__ Wzt) {
    int i = blockIdx.x * blockDim.x + threadIdx.x;
    if (i >= ADIM * DM) return;
    int ha = i & (DM - 1);
    int a  = i >> 10;
    Wzt[i] = (_Float16)Wz[(size_t)ha * ADIM + a];
}

// ---------------------------------------------------------------------------
// QKV projection, async-LDS double-buffered, batched B fragments.
// Block = 8 waves = 128-row M-tile x one head.
// ---------------------------------------------------------------------------
#define AROW_PAD 40                         // halves per LDS row (80 B)
#define A_ROWS   128
#define B_ROWS   (3 * ADIM)                 // 192
#define A_STAGE_H (A_ROWS * AROW_PAD)
#define B_STAGE_H (B_ROWS * AROW_PAD)
#define STAGE_H  (A_STAGE_H + B_STAGE_H)    // 12800 halves = 25600 B

__global__ void __launch_bounds__(256)
qkv_kernel(const _Float16* __restrict__ x16, const _Float16* __restrict__ Wt,
           _Float16* __restrict__ q, _Float16* __restrict__ k, _Float16* __restrict__ vT) {
    __shared__ alignas(16) _Float16 smem[2 * STAGE_H];  // 51200 B

    int tid  = threadIdx.x;
    int wv   = tid >> 5;
    int lane = tid & 31;
    int m = lane & 15, hw = lane >> 4;

    int h        = blockIdx.x & (HEADS - 1);
    int row0_blk = (blockIdx.x >> 4) * A_ROWS;
    int b        = row0_blk / SEQ_;
    int bh       = b * HEADS + h;

    const _Float16* xrow = x16 + (size_t)row0_blk * DM;
    const _Float16* wh   = Wt + (size_t)h * 3 * ADIM * DM;

    auto fill = [&](int stage, int ks) {
        _Float16* sA = smem + stage * STAGE_H;
        _Float16* sB = sA + A_STAGE_H;
        for (int idx = tid; idx < A_ROWS * 4; idx += 256) {
            int r = idx >> 2, c = idx & 3;
            async_copy16((unsigned)(uintptr_t)(sA + r * AROW_PAD + c * 8),
                         xrow + (size_t)r * DM + ks + c * 8);
        }
        for (int idx = tid; idx < B_ROWS * 4; idx += 256) {
            int r = idx >> 2, c = idx & 3;
            async_copy16((unsigned)(uintptr_t)(sB + r * AROW_PAD + c * 8),
                         wh + (size_t)r * DM + ks + c * 8);
        }
    };

    floatx8 acc[3][4] = {};

    fill(0, 0);
    wait_async0();
    __syncthreads();

    for (int step = 0; step < DM / 32; ++step) {
        if (step + 1 < DM / 32) fill((step + 1) & 1, (step + 1) * 32);

        const _Float16* sA = smem + (step & 1) * STAGE_H;
        const _Float16* sB = sA + A_STAGE_H;

        auto loadB12 = [&](int i) {     // i = p*4 + j
            return load_b(sB + (i >> 2) * ADIM * AROW_PAD, AROW_PAD, (i & 3) * 16, 0, m, hw);
        };

        half16 afrag = load_a(sA, AROW_PAD, wv * 16, 0, m, hw);
#pragma unroll
        for (int g = 0; g < 2; ++g) {   // two batches of 6: loads up front, fence, wmma burst
            half16 bf[6];
#pragma unroll
            for (int i = 0; i < 6; ++i) bf[i] = loadB12(g * 6 + i);
            sched_fence();
#pragma unroll
            for (int i = 0; i < 6; ++i) {
                int t = g * 6 + i;
                acc[t >> 2][t & 3] = wmma_f16(afrag, bf[i], acc[t >> 2][t & 3]);
            }
        }
        wait_async0();
        __syncthreads();
    }

    int row0    = row0_blk + wv * 16;
    int s_local = row0 - b * SEQ_;
    size_t qbase = (size_t)bh * SEQ_ * ADIM;
#pragma unroll
    for (int j = 0; j < 4; ++j) {
#pragma unroll
        for (int r = 0; r < 8; ++r) {
            int srow = s_local + r + hw * 8;
            int col  = j * 16 + m;
            q[qbase + (size_t)srow * ADIM + col] = (_Float16)acc[0][j][r];
            k[qbase + (size_t)srow * ADIM + col] = (_Float16)acc[1][j][r];
            vT[(size_t)bh * ADIM * SEQ_ + (size_t)col * SEQ_ + srow] = (_Float16)acc[2][j][r];
        }
    }
}

// ---------------------------------------------------------------------------
// Flash attention. Block = 4 waves sharing one (b,h); async-LDS k/v tiles.
// ---------------------------------------------------------------------------
#define KVPAD 72                            // halves per LDS row (144 B)

__global__ void __launch_bounds__(128)
attn_kernel(const _Float16* __restrict__ q, const _Float16* __restrict__ k,
            const _Float16* __restrict__ vT, _Float16* __restrict__ cat) {
    __shared__ alignas(16) _Float16 kvbuf[2][2][64 * KVPAD];  // 36864 B
    __shared__ alignas(16) _Float16 pbuf[4][16 * 64];         // 8192 B

    int tid   = threadIdx.x;
    int wslot = tid >> 5;
    int wave  = blockIdx.x * 4 + wslot;
    int lane  = tid & 31;
    int m = lane & 15, hw = lane >> 4;

    int bh    = wave >> 6;
    int stile = wave & 63;
    int s0    = stile * 16;

    const _Float16* qp = q  + (size_t)bh * SEQ_ * ADIM;
    const _Float16* kp = k  + (size_t)bh * SEQ_ * ADIM;
    const _Float16* vp = vT + (size_t)bh * ADIM * SEQ_;
    _Float16* pb = pbuf[wslot];

    auto fillkv = [&](int stage, int t) {
        _Float16* sk = kvbuf[stage][0];
        _Float16* sv = kvbuf[stage][1];
        for (int idx = tid; idx < 64 * 8; idx += 128) {
            int r = idx >> 3, c = idx & 7;
            async_copy16((unsigned)(uintptr_t)(sk + r * KVPAD + c * 8),
                         kp + (size_t)(t + r) * ADIM + c * 8);
        }
        for (int idx = tid; idx < 64 * 8; idx += 128) {
            int r = idx >> 3, c = idx & 7;
            async_copy16((unsigned)(uintptr_t)(sv + r * KVPAD + c * 8),
                         vp + (size_t)r * SEQ_ + t + c * 8);
        }
    };

    half16 qa0 = load_a(qp, ADIM, s0, 0,  m, hw);
    half16 qa1 = load_a(qp, ADIM, s0, 32, m, hw);

    floatx8 o[4] = {};
    float mrow[8], lrow[8];
#pragma unroll
    for (int r = 0; r < 8; ++r) { mrow[r] = -1e30f; lrow[r] = 0.0f; }

    fillkv(0, 0);
    wait_async0();
    __syncthreads();

    for (int t = 0, step = 0; t < SEQ_; t += 64, ++step) {
        if (t + 64 < SEQ_) fillkv((step + 1) & 1, t + 64);

        const _Float16* sk = kvbuf[step & 1][0];
        const _Float16* sv = kvbuf[step & 1][1];

        // ---- scores: batch all 8 k-fragments, fence, 8-WMMA burst ----
        floatx8 sc[4] = {};
        {
            half16 kf[8];
#pragma unroll
            for (int i = 0; i < 8; ++i)        // i = tt*2 + khalf
                kf[i] = load_b(sk, KVPAD, (i >> 1) * 16, (i & 1) * 32, m, hw);
            sched_fence();
#pragma unroll
            for (int i = 0; i < 8; ++i)
                sc[i >> 1] = wmma_f16((i & 1) ? qa1 : qa0, kf[i], sc[i >> 1]);
        }

        // ---- online softmax update ----
        float alpha[8];
#pragma unroll
        for (int r = 0; r < 8; ++r) {
            float mx = -1e30f;
#pragma unroll
            for (int tt = 0; tt < 4; ++tt) {
                sc[tt][r] *= 0.125f;                 // 1/sqrt(64)
                mx = fmaxf(mx, sc[tt][r]);
            }
#pragma unroll
            for (int d = 1; d < 16; d <<= 1) mx = fmaxf(mx, __shfl_xor(mx, d, 16));
            float mnew = fmaxf(mrow[r], mx);
            alpha[r] = __expf(mrow[r] - mnew);
            mrow[r]  = mnew;

            float ps = 0.0f;
#pragma unroll
            for (int tt = 0; tt < 4; ++tt) {
                float p = __expf(sc[tt][r] - mnew);
                ps += p;
                pb[(r + hw * 8) * 64 + tt * 16 + m] = (_Float16)p;  // C-layout -> LDS
            }
#pragma unroll
            for (int d = 1; d < 16; d <<= 1) ps += __shfl_xor(ps, d, 16);
            lrow[r] = lrow[r] * alpha[r] + ps;
        }

#pragma unroll
        for (int j = 0; j < 4; ++j)
#pragma unroll
            for (int r = 0; r < 8; ++r) o[j][r] *= alpha[r];

        __builtin_amdgcn_wave_barrier();     // order probs stores -> A-frag loads

        // ---- PV: batch probs A-frags + 8 v-fragments, fence, 8-WMMA burst ----
        {
            half16 pa0 = load_a(pb, 64, 0, 0,  m, hw);
            half16 pa1 = load_a(pb, 64, 0, 32, m, hw);
            half16 vf[8];
#pragma unroll
            for (int i = 0; i < 8; ++i)        // i = j*2 + khalf
                vf[i] = load_b(sv, KVPAD, (i >> 1) * 16, (i & 1) * 32, m, hw);
            sched_fence();
#pragma unroll
            for (int i = 0; i < 8; ++i)
                o[i >> 1] = wmma_f16((i & 1) ? pa1 : pa0, vf[i], o[i >> 1]);
        }

        wait_async0();
        __syncthreads();
    }

    int b_ = bh / HEADS, h_ = bh % HEADS;
#pragma unroll
    for (int j = 0; j < 4; ++j)
#pragma unroll
        for (int r = 0; r < 8; ++r) {
            int   srow = s0 + r + hw * 8;
            float val  = o[j][r] / lrow[r];
            cat[((size_t)b_ * SEQ_ + srow) * DM + h_ * ADIM + j * 16 + m] = (_Float16)val;
        }
}

// ---------------------------------------------------------------------------
// Output projection: out[8192][64] f32 = cat[8192][1024] x Wz[1024][64]
// ---------------------------------------------------------------------------
__global__ void __launch_bounds__(128)
outproj_kernel(const _Float16* __restrict__ cat, const _Float16* __restrict__ Wzt,
               float* __restrict__ out) {
    int wave = blockIdx.x * (blockDim.x >> 5) + (threadIdx.x >> 5);
    int lane = threadIdx.x & 31;
    int m = lane & 15, hw = lane >> 4;
    int row0 = wave * 16;

    floatx8 acc[4] = {};
    for (int ks = 0; ks < DM; ks += 32) {
        half16 afrag = load_a(cat, DM, row0, ks, m, hw);
        half16 bf[4];
#pragma unroll
        for (int j = 0; j < 4; ++j) bf[j] = load_b(Wzt, DM, j * 16, ks, m, hw);
        sched_fence();
#pragma unroll
        for (int j = 0; j < 4; ++j) acc[j] = wmma_f16(afrag, bf[j], acc[j]);
    }
#pragma unroll
    for (int j = 0; j < 4; ++j)
#pragma unroll
        for (int r = 0; r < 8; ++r)
            out[(size_t)(row0 + r + hw * 8) * ADIM + j * 16 + m] = acc[j][r];
}

// ---------------------------------------------------------------------------
extern "C" void kernel_launch(void* const* d_in, const int* in_sizes, int n_in,
                              void* d_out, int out_size, void* d_ws, size_t ws_size,
                              hipStream_t stream) {
    const float* x  = (const float*)d_in[0];   // [8,1024,1024]
    const float* W  = (const float*)d_in[1];   // [16,3,1024,64]
    const float* Wz = (const float*)d_in[2];   // [1024,64]
    float* out = (float*)d_out;                // [8,1024,64]

    char* ws = (char*)d_ws;
    size_t off = 0;
    _Float16* x16 = (_Float16*)(ws + off); off += (size_t)ROWS * DM * 2;
    _Float16* Wt  = (_Float16*)(ws + off); off += (size_t)HEADS * 3 * ADIM * DM * 2;
    _Float16* Wzt = (_Float16*)(ws + off); off += (size_t)ADIM * DM * 2;
    _Float16* qb  = (_Float16*)(ws + off); off += (size_t)BATCH_ * HEADS * SEQ_ * ADIM * 2;
    _Float16* kb  = (_Float16*)(ws + off); off += (size_t)BATCH_ * HEADS * SEQ_ * ADIM * 2;
    _Float16* vT  = (_Float16*)(ws + off); off += (size_t)BATCH_ * HEADS * ADIM * SEQ_ * 2;
    _Float16* cat = (_Float16*)(ws + off); off += (size_t)ROWS * DM * 2;

    {
        int n = ROWS * DM;
        cvt_x_kernel<<<(n + 255) / 256, 256, 0, stream>>>(x, x16, n);
    }
    {
        int n = HEADS * 3 * ADIM * DM;
        transpose_w_kernel<<<(n + 255) / 256, 256, 0, stream>>>(W, Wt);
    }
    {
        int n = ADIM * DM;
        transpose_wz_kernel<<<(n + 255) / 256, 256, 0, stream>>>(Wz, Wzt);
    }

    qkv_kernel<<<(ROWS / A_ROWS) * HEADS, 256, 0, stream>>>(x16, Wt, qb, kb, vT);
    attn_kernel<<<BATCH_ * HEADS * (SEQ_ / 16) / 4, 128, 0, stream>>>(qb, kb, vT, cat);
    outproj_kernel<<<(ROWS / 16) / 4, 128, 0, stream>>>(cat, Wzt, out);
}
